// RectConv2d_71012989272744
// MI455X (gfx1250) — matrix operand order repaired and checked by reading, not compiled, not run on hardware
//
#include <hip/hip_runtime.h>
#include <math.h>

typedef _Float16 f16_t;
typedef __attribute__((ext_vector_type(16))) _Float16 v16h;
typedef __attribute__((ext_vector_type(8)))  _Float16 v8h;
typedef __attribute__((ext_vector_type(8)))  float    v8f;
typedef int v4i32 __attribute__((vector_size(16)));
typedef __attribute__((address_space(1))) v4i32* gvec_p;
typedef __attribute__((address_space(3))) v4i32* lvec_p;

#define BATCH 4
#define CH    64
#define HH    128
#define WW    128
#define NPOS  (BATCH*HH*WW)     /* 65536 */
#define KTOT  576               /* 9 taps * 64 ch */
#define KC    18                /* K chunks of 32 */
#define APSZ  (KC*4*32*16)      /* halves per packed weight tensor = 36864 (72KB) */

union FragH { v16h v; v8h h2[2]; f16_t f[16]; };
union H8    { v8h  v; f16_t f[8]; };

// ---------------- async global -> LDS staging of packed weights ----------------
__device__ __forceinline__ void cp16_g2l(const f16_t* g, f16_t* l) {
#if defined(__gfx1250__) && __has_builtin(__builtin_amdgcn_global_load_async_to_lds_b128)
  __builtin_amdgcn_global_load_async_to_lds_b128((gvec_p)g, (lvec_p)l, 0, 0);
#else
  *(v8h*)l = *(const v8h*)g;
#endif
}

__device__ __forceinline__ void cp_wait_barrier() {
#if defined(__gfx1250__) && __has_builtin(__builtin_amdgcn_global_load_async_to_lds_b128)
#if __has_builtin(__builtin_amdgcn_s_wait_asynccnt)
  __builtin_amdgcn_s_wait_asynccnt(0);
#else
  asm volatile("s_wait_asynccnt 0x0" ::: "memory");
#endif
#endif
  __syncthreads();
}

__device__ __forceinline__ void stage_A(const f16_t* __restrict__ gA,
                                        f16_t* __restrict__ sA) {
  // 36864 halves = 4608 x (8 halves); 256 threads x 18 chunks
  const int tid = threadIdx.x;
#pragma unroll
  for (int i = 0; i < 18; ++i) {
    const int off = (i * 256 + tid) * 8;
    cp16_g2l(gA + off, sA + off);
  }
  cp_wait_barrier();
}

// ---------------- NCHW f32 -> NHWC f16 ----------------
__global__ __launch_bounds__(256) void k_cvt(const float* __restrict__ x,
                                             f16_t* __restrict__ xh) {
  int t = blockIdx.x * 256 + threadIdx.x;           // exactly B*C*H*W threads
  int w = t & 127, h = (t >> 7) & 127, c = (t >> 14) & 63, b = t >> 20;
  xh[(((size_t)((b * HH + h) * WW + w)) << 6) + c] = (f16_t)x[t];
}

// ---------------- pack OIHW f32 weights into WMMA A-fragment layout ----------
// dst[((kc*4+mt)*32 + lane)*16 + e] = W[oc = mt*16+lane%16][K = kc*32 + kA(e,lane)]
// K = tap*64 + c ; weight flat index = (oc*64 + c)*9 + tap
__global__ __launch_bounds__(256) void k_pack(const float* __restrict__ wsrc,
                                              f16_t* __restrict__ ap) {
  int t = blockIdx.x * 256 + threadIdx.x;           // exactly 18*4*32 = 2304
  int lane = t & 31;
  int mt   = (t >> 5) & 3;
  int kc   = t >> 7;
  int m    = mt * 16 + (lane & 15);
  int hi8  = (lane >= 16) ? 8 : 0;
  f16_t* dst = ap + (size_t)t * 16;
#pragma unroll
  for (int e = 0; e < 16; ++e) {
    int k = kc * 32 + e + 2 * hi8;
    int s = k >> 6;
    int c = k & 63;
    dst[e] = (f16_t)wsrc[((size_t)(m * 64 + c)) * 9 + s];
  }
}

#define WMMA(ACC, AV, BV) \
  ACC = __builtin_amdgcn_wmma_f32_16x16x32_f16(false, AV, false, BV, (short)0, ACC, false, false)

#define LOAD_A4(ap, A0, A1, A2, A3)                                         \
  A0.h2[0] = *(const v8h*)(ap);        A0.h2[1] = *(const v8h*)(ap + 8);    \
  A1.h2[0] = *(const v8h*)(ap + 512);  A1.h2[1] = *(const v8h*)(ap + 520);  \
  A2.h2[0] = *(const v8h*)(ap + 1024); A2.h2[1] = *(const v8h*)(ap + 1032); \
  A3.h2[0] = *(const v8h*)(ap + 1536); A3.h2[1] = *(const v8h*)(ap + 1544)

// ---------------- 3x3 pad-1 conv, implicit GEMM via WMMA ----------------
// act: 0 none, 1 relu, 2 tanh ; writes NHWC f16
__global__ __launch_bounds__(256) void k_conv3x3(
    const f16_t* __restrict__ src, const f16_t* __restrict__ gA,
    const float* __restrict__ bias, int act, f16_t* __restrict__ outh) {
  __shared__ f16_t sA[APSZ];
  stage_A(gA, sA);

  const int lane = threadIdx.x & 31;
  const int wave = threadIdx.x >> 5;
  const int nt   = blockIdx.x * 8 + wave;   // 0..4095
  const int lan  = lane & 15;
  const int hi   = lane >> 4;
  const int n    = (nt << 4) + lan;
  const int w    = n & 127;
  const int h    = (n >> 7) & 127;
  const int b    = n >> 14;

  v8f z8 = {};
  v8f acc0 = z8, acc1 = z8, acc2 = z8, acc3 = z8;

#pragma unroll
  for (int s = 0; s < 9; ++s) {
    const int dh = s / 3 - 1, dw = s % 3 - 1;   // constants after unroll
    const int hh2 = h + dh;
    if ((unsigned)hh2 < HH) {                   // wave-uniform row check
      int wc = w + dw;
      const bool okw = (unsigned)wc < WW;       // folds to true when dw==0
      wc = wc < 0 ? 0 : (wc > WW - 1 ? WW - 1 : wc);
      const f16_t* p = src + ((((size_t)b * HH + hh2) * WW + wc) << 6) + (hi << 4);
      // B fragments: chunks kc=2s (c0..31), 2s+1 (c32..63)
      FragH b0, b1;
      b0.h2[0] = *(const v8h*)(p);
      b0.h2[1] = *(const v8h*)(p + 8);
      b1.h2[0] = *(const v8h*)(p + 32);
      b1.h2[1] = *(const v8h*)(p + 40);
      if (!okw) {
        v8h zh = {};
        b0.h2[0] = zh; b0.h2[1] = zh; b1.h2[0] = zh; b1.h2[1] = zh;
      }
      // all 8 A fragments of this tap up-front (distinct regs, one big clause)
      const f16_t* ap0 = sA + (2 * s) * 2048 + lane * 16;
      const f16_t* ap1 = ap0 + 2048;
      FragH A0, A1, A2, A3, A4, A5, A6, A7;
      LOAD_A4(ap0, A0, A1, A2, A3);
      LOAD_A4(ap1, A4, A5, A6, A7);
      WMMA(acc0, A0.v, b0.v);
      WMMA(acc1, A1.v, b0.v);
      WMMA(acc2, A2.v, b0.v);
      WMMA(acc3, A3.v, b0.v);
      WMMA(acc0, A4.v, b1.v);
      WMMA(acc1, A5.v, b1.v);
      WMMA(acc2, A6.v, b1.v);
      WMMA(acc3, A7.v, b1.v);
    }
  }

  v8f accs[4] = {acc0, acc1, acc2, acc3};
#pragma unroll
  for (int mt = 0; mt < 4; ++mt) {
#pragma unroll
    for (int v = 0; v < 8; ++v) {
      const int oc = mt * 16 + v + (hi << 3);
      float val = accs[mt][v] + bias[oc];
      if (act == 1)      val = fmaxf(val, 0.f);
      else if (act == 2) val = tanhf(val);
      outh[((size_t)n << 6) + oc] = (f16_t)val;
    }
  }
}

// ---------------- final conv: K-contiguous sampled tensor, fused *m + bias ----
__global__ __launch_bounds__(256) void k_convflat(
    const f16_t* __restrict__ xs, const f16_t* __restrict__ gA,
    const float* __restrict__ bias, const f16_t* __restrict__ mT,
    const f16_t* __restrict__ bT, float* __restrict__ outf) {
  __shared__ f16_t sA[APSZ];
  stage_A(gA, sA);

  const int lane = threadIdx.x & 31;
  const int wave = threadIdx.x >> 5;
  const int nt   = blockIdx.x * 8 + wave;
  const int lan  = lane & 15;
  const int hi   = lane >> 4;
  const int n    = (nt << 4) + lan;
  const int w    = n & 127;
  const int h    = (n >> 7) & 127;
  const int b    = n >> 14;

  v8f z8 = {};
  v8f acc0 = z8, acc1 = z8, acc2 = z8, acc3 = z8;
  const f16_t* base = xs + (size_t)n * KTOT + (hi << 4);

#pragma unroll
  for (int kp = 0; kp < 9; ++kp) {            // pairs of K chunks
    FragH b0, b1;
    b0.h2[0] = *(const v8h*)(base + (2 * kp) * 32);
    b0.h2[1] = *(const v8h*)(base + (2 * kp) * 32 + 8);
    b1.h2[0] = *(const v8h*)(base + (2 * kp + 1) * 32);
    b1.h2[1] = *(const v8h*)(base + (2 * kp + 1) * 32 + 8);
    const f16_t* ap0 = sA + (2 * kp) * 2048 + lane * 16;
    const f16_t* ap1 = ap0 + 2048;
    FragH A0, A1, A2, A3, A4, A5, A6, A7;
    LOAD_A4(ap0, A0, A1, A2, A3);
    LOAD_A4(ap1, A4, A5, A6, A7);
    WMMA(acc0, A0.v, b0.v);
    WMMA(acc1, A1.v, b0.v);
    WMMA(acc2, A2.v, b0.v);
    WMMA(acc3, A3.v, b0.v);
    WMMA(acc0, A4.v, b1.v);
    WMMA(acc1, A5.v, b1.v);
    WMMA(acc2, A6.v, b1.v);
    WMMA(acc3, A7.v, b1.v);
  }

  v8f accs[4] = {acc0, acc1, acc2, acc3};
#pragma unroll
  for (int mt = 0; mt < 4; ++mt) {
#pragma unroll
    for (int v = 0; v < 8; ++v) {
      const int oc = mt * 16 + v + (hi << 3);
      const size_t ni = ((size_t)n << 6) + oc;
      float val = accs[mt][v] + bias[oc];
      float mv = (float)mT[ni];
      float bv = (float)bT[ni];
      outf[(((size_t)b * 64 + oc) << 14) + (h << 7) + w] = val * mv + bv;
    }
  }
}

// ---------------- l / wd scale fields ----------------
__global__ __launch_bounds__(256) void k_lwd(
    const f16_t* __restrict__ off, const float* __restrict__ lw1,
    const float* __restrict__ lb1, const float* __restrict__ lw2,
    const float* __restrict__ lb2, const float* __restrict__ ww1,
    const float* __restrict__ wb1, const float* __restrict__ ww2,
    const float* __restrict__ wb2, float* __restrict__ lout,
    float* __restrict__ wout) {
  int n = blockIdx.x * 256 + threadIdx.x;            // exactly 65536
  int w = n & 127, h = (n >> 7) & 127, b = n >> 14;
  float tl = 0.f, tw = 0.f;
  for (int kh = 0; kh < 3; ++kh)
    for (int kw2 = 0; kw2 < 3; ++kw2) {
      int hh2 = h + kh - 1, ww2 = w + kw2 - 1;
      if ((unsigned)hh2 < HH && (unsigned)ww2 < WW) {
        const f16_t* p = off + ((((size_t)b * HH + hh2) * WW + ww2) << 6);
        for (int c = 0; c < 64; ++c) {
          float v = (float)p[c];
          tl += v * lw1[(c * 3 + kh) * 3 + kw2];
          tw += v * ww1[(c * 3 + kh) * 3 + kw2];
        }
      }
    }
  tl = fmaxf(tl + lb1[0], 0.f);
  tw = fmaxf(tw + wb1[0], 0.f);
  lout[n] = 1.f / (1.f + expf(-(tl * lw2[0] + lb2[0]))) * 8.f + 1.f;
  wout[n] = 1.f / (1.f + expf(-(tw * ww2[0] + wb2[0]))) * 8.f + 1.f;
}

__device__ __forceinline__ const f16_t* cptr(const f16_t* xh, int b, int i, int j) {
  if (i >= 1 && i <= HH && j >= 1 && j <= WW)
    return xh + ((((size_t)b * HH + (i - 1)) * WW + (j - 1)) << 6);
  return nullptr;
}

// ---------------- deformable bilinear sampler -> xs[pos][tap][C] f16 ----------
__global__ __launch_bounds__(256) void k_sample(
    const f16_t* __restrict__ xh, const float* __restrict__ lbuf,
    const float* __restrict__ wbuf, f16_t* __restrict__ xs) {
  int t = blockIdx.x * 256 + threadIdx.x;            // exactly 589824
  int s = t % 9;
  int n = t / 9;
  int w = n & 127, h = (n >> 7) & 127, b = n >> 14;
  float Lr = lbuf[n] * (1.f / 3.f);
  float Wr = wbuf[n] * (1.f / 3.f);
  float pnx = (float)(s / 3 - 1);
  float pny = (float)(s % 3 - 1);
  float px = (float)(h + 1) + Lr * pnx;
  float py = (float)(w + 1) + Wr * pny;
  float fx = floorf(px), fy = floorf(py);
  float pxc = fminf(fmaxf(px, 0.f), 129.f);
  float pyc = fminf(fmaxf(py, 0.f), 129.f);
  int qxl = (int)fminf(fmaxf(fx, 0.f), 129.f);
  int qyl = (int)fminf(fmaxf(fy, 0.f), 129.f);
  int qxr = (int)fminf(fmaxf(fx + 1.f, 0.f), 129.f);
  int qyr = (int)fminf(fmaxf(fy + 1.f, 0.f), 129.f);
  float glt = (1.f + ((float)qxl - pxc)) * (1.f + ((float)qyl - pyc));
  float grb = (1.f - ((float)qxr - pxc)) * (1.f - ((float)qyr - pyc));
  float glb = (1.f + ((float)qxl - pxc)) * (1.f - ((float)qyr - pyc));
  float grt = (1.f - ((float)qxr - pxc)) * (1.f + ((float)qyl - pyc));
  const f16_t* plt = cptr(xh, b, qxl, qyl);
  const f16_t* prb = cptr(xh, b, qxr, qyr);
  const f16_t* plb = cptr(xh, b, qxl, qyr);
  const f16_t* prt = cptr(xh, b, qxr, qyl);
  f16_t* dst = xs + (size_t)t * 64;
  H8 zv = {};
  for (int c = 0; c < 64; c += 8) {
    H8 a, bb, cc, dd, o;
    a.v  = plt ? *(const v8h*)(plt + c) : zv.v;
    bb.v = prb ? *(const v8h*)(prb + c) : zv.v;
    cc.v = plb ? *(const v8h*)(plb + c) : zv.v;
    dd.v = prt ? *(const v8h*)(prt + c) : zv.v;
#pragma unroll
    for (int e = 0; e < 8; ++e)
      o.f[e] = (f16_t)(glt * (float)a.f[e] + grb * (float)bb.f[e] +
                       glb * (float)cc.f[e] + grt * (float)dd.f[e]);
    *(v8h*)(dst + c) = o.v;
  }
}

extern "C" void kernel_launch(void* const* d_in, const int* in_sizes, int n_in,
                              void* d_out, int out_size, void* d_ws, size_t ws_size,
                              hipStream_t stream) {
  (void)in_sizes; (void)n_in; (void)out_size; (void)ws_size;
  const float* x   = (const float*)d_in[0];
  const float* bw1 = (const float*)d_in[1];  const float* bb1 = (const float*)d_in[2];
  const float* bw2 = (const float*)d_in[3];  const float* bb2 = (const float*)d_in[4];
  const float* mw1 = (const float*)d_in[5];  const float* mb1 = (const float*)d_in[6];
  const float* mw2 = (const float*)d_in[7];  const float* mb2 = (const float*)d_in[8];
  const float* pw1 = (const float*)d_in[9];  const float* pb1 = (const float*)d_in[10];
  const float* pw2 = (const float*)d_in[11]; const float* pb2 = (const float*)d_in[12];
  const float* lw1 = (const float*)d_in[13]; const float* lb1 = (const float*)d_in[14];
  const float* lw2 = (const float*)d_in[15]; const float* lb2 = (const float*)d_in[16];
  const float* ww1 = (const float*)d_in[17]; const float* wb1 = (const float*)d_in[18];
  const float* ww2 = (const float*)d_in[19]; const float* wb2 = (const float*)d_in[20];
  const float* kw  = (const float*)d_in[21]; const float* kb  = (const float*)d_in[22];

  char* ws = (char*)d_ws;
  const size_t MB = 1u << 20;
  f16_t* xh    = (f16_t*)(ws + 0 * MB);    // 8 MB  NHWC f16 input
  f16_t* t1    = (f16_t*)(ws + 8 * MB);    // 8 MB  reused first-layer relu output
  f16_t* biasT = (f16_t*)(ws + 16 * MB);   // 8 MB
  f16_t* mT    = (f16_t*)(ws + 24 * MB);   // 8 MB (already tanh'd)
  f16_t* offT  = (f16_t*)(ws + 32 * MB);   // 8 MB
  float* lbuf  = (float*)(ws + 40 * MB);   // 256 KB
  float* wbuf  = (float*)(ws + 41 * MB);   // 256 KB
  f16_t* apack = (f16_t*)(ws + 42 * MB);   // 7 * 72 KB packed weights
  f16_t* xs    = (f16_t*)(ws + 43 * MB);   // 72 MB sampled tensor [pos][tap][C]

  k_cvt<<<(BATCH * CH * HH * WW) / 256, 256, 0, stream>>>(x, xh);

  const float* wsrcs[7] = {bw1, bw2, mw1, mw2, pw1, pw2, kw};
  for (int i = 0; i < 7; ++i)
    k_pack<<<9, 256, 0, stream>>>(wsrcs[i], apack + (size_t)i * APSZ);

  // bias branch
  k_conv3x3<<<512, 256, 0, stream>>>(xh, apack + 0 * APSZ, bb1, 1, t1);
  k_conv3x3<<<512, 256, 0, stream>>>(t1, apack + 1 * APSZ, bb2, 0, biasT);
  // m branch (tanh fused)
  k_conv3x3<<<512, 256, 0, stream>>>(xh, apack + 2 * APSZ, mb1, 1, t1);
  k_conv3x3<<<512, 256, 0, stream>>>(t1, apack + 3 * APSZ, mb2, 2, mT);
  // offset branch
  k_conv3x3<<<512, 256, 0, stream>>>(xh, apack + 4 * APSZ, pb1, 1, t1);
  k_conv3x3<<<512, 256, 0, stream>>>(t1, apack + 5 * APSZ, pb2, 0, offT);

  k_lwd<<<256, 256, 0, stream>>>(offT, lw1, lb1, lw2, lb2, ww1, wb1, ww2, wb2, lbuf, wbuf);
  k_sample<<<2304, 256, 0, stream>>>(xh, lbuf, wbuf, xs);

  // final conv over K=576 sampled values, fused *m + bias, NCHW f32 to d_out
  k_convflat<<<512, 256, 0, stream>>>(xs, apack + 6 * APSZ, kb, mT, biasT, (float*)d_out);
}
